// LocalFeatureAggregation_34239479284328
// MI455X (gfx1250) — matrix operand order repaired and checked by reading, not compiled, not run on hardware
//
#include <hip/hip_runtime.h>
#include <math.h>

// ---------------------------------------------------------------------------
// RandLA-Net LocalFeatureAggregation on gfx1250 (MI455X), WMMA f16 pipeline.
// B=2, N=32768, K=16, CIN=8, COUT=64.  Compute-bound (~220 flop/byte), so all
// per-neighbor GEMMs are mapped to v_wmma_f32_16x16x32_f16 with the 16
// neighbors of one point forming the M=16 rows of each tile.
// ---------------------------------------------------------------------------

typedef __attribute__((ext_vector_type(16))) _Float16 v16h;
typedef __attribute__((ext_vector_type(8)))  float    v8f;

#define NPTS   32768
#define KNBR   16
#define CIN    8
#define COUT   64
#define BN_INV 0.9999950000374996f   // 1/sqrt(1 + 1e-5)

// sW layout (floats): lse_w1[0..31] lse_g1[32..39] lse_b1[40..47]
// lse_w2[48..175] lse_g2[176..191] lse_b2[192..207] m1_g[208..239]
// m1_b[240..271] m2_g[272..335] m2_b[336..399] pool_g[400..463] pool_b[464..527]

__device__ __forceinline__ v8f wmma_f16(v16h a, v16h b, v8f c) {
  // D = A(16x32 f16) * B(32x16 f16) + C(16x16 f32)
  return __builtin_amdgcn_wmma_f32_16x16x32_f16(false, a, false, b,
                                                (short)0, c, false, false);
}

// B-fragment of B = W^T for one 32x16 tile.  W is row-major (OUT x IN).
// B[k][n] = W[n][k]; zero pad for k >= IN.  Layout: lanes 0-15 hold K=k0..k0+7,
// k0+16..k0+23 of column n0+lane; lanes 16-31 hold K=k0+8..k0+15, k0+24..k0+31.
__device__ __forceinline__ v16h load_btile(const float* __restrict__ W, int IN,
                                           int n0, int k0, int lane) {
  v16h b;
  const int n  = n0 + (lane & 15);
  const int kb = k0 + ((lane >> 4) << 3);
#pragma unroll
  for (int i = 0; i < 16; ++i) {
    const int k = kb + i + ((i >> 3) << 3);
    const float v = (k < IN) ? W[n * IN + k] : 0.0f;
    b[i] = (_Float16)v;
  }
  return b;
}

__global__ __launch_bounds__(128)
void lfa_wmma_kernel(const float* __restrict__ coords,
                     const float* __restrict__ features,
                     const int*   __restrict__ nbr,
                     const float* __restrict__ lse_w1,
                     const float* __restrict__ lse_g1,
                     const float* __restrict__ lse_b1,
                     const float* __restrict__ lse_w2,
                     const float* __restrict__ lse_g2,
                     const float* __restrict__ lse_b2,
                     const float* __restrict__ m1_w,
                     const float* __restrict__ m1_g,
                     const float* __restrict__ m1_b,
                     const float* __restrict__ m2_w,
                     const float* __restrict__ m2_g,
                     const float* __restrict__ m2_b,
                     const float* __restrict__ att_w,
                     const float* __restrict__ pool_w,
                     const float* __restrict__ pool_g,
                     const float* __restrict__ pool_b,
                     float* __restrict__ out) {
  __shared__ _Float16 sX[4][16][64];     // per-wave layer transpose staging
  __shared__ float    sPooled[16][64];   // pooled features for 16 points
  __shared__ float    sW[528];           // small per-channel weights

  const int tid  = threadIdx.x;
  const int lane = tid & 31;
  const int wave = tid >> 5;
  const int col  = lane & 15;            // C-tile column / neighbor row id
  const int hi   = lane >> 4;            // half-wave selector
  const int m0   = hi * 8;               // first C-tile row held by this lane

  // ---- stage small weights into LDS -------------------------------------
  if (tid < 32)  sW[0 + tid]   = lse_w1[tid];
  if (tid < 8)  { sW[32 + tid] = lse_g1[tid]; sW[40 + tid] = lse_b1[tid]; }
  if (tid < 128) sW[48 + tid]  = lse_w2[tid];
  if (tid < 16) { sW[176 + tid] = lse_g2[tid]; sW[192 + tid] = lse_b2[tid]; }
  if (tid < 32) { sW[208 + tid] = m1_g[tid];   sW[240 + tid] = m1_b[tid]; }
  if (tid < 64) { sW[272 + tid] = m2_g[tid];   sW[336 + tid] = m2_b[tid];
                  sW[400 + tid] = pool_g[tid]; sW[464 + tid] = pool_b[tid]; }
  __syncthreads();

  // ---- resident f16 B-fragments (weights transposed into WMMA layout) ---
  v16h Bm1[2], Bm2[4], Ba[2][4];
#pragma unroll
  for (int t = 0; t < 2; ++t) Bm1[t] = load_btile(m1_w, 24, t * 16, 0, lane);
#pragma unroll
  for (int t = 0; t < 4; ++t) Bm2[t] = load_btile(m2_w, 32, t * 16, 0, lane);
#pragma unroll
  for (int kc = 0; kc < 2; ++kc)
#pragma unroll
    for (int t = 0; t < 4; ++t)
      Ba[kc][t] = load_btile(att_w, 64, t * 16, kc * 32, lane);

  const v8f zero = {};

  // ---- phase 1: each wave handles 4 points (16 neighbors = 16 M-rows) ---
  for (int it = 0; it < 4; ++it) {
    const int p  = blockIdx.x * 16 + wave * 4 + it;  // flattened (b, n)
    const int bb = p >> 15;                          // p / NPTS
    const int r  = col;                              // neighbor row id

    // gather neighbor + local spatial encoding (VALU; both halves redundant)
    const int idx = nbr[(size_t)p * KNBR + r];
    const float* cc = coords + (size_t)p * 3;
    const float* cn = coords + ((size_t)bb * NPTS + idx) * 3;
    const float rx = cn[0] - cc[0];
    const float ry = cn[1] - cc[1];
    const float rz = cn[2] - cc[2];
    const float dd = sqrtf(rx * rx + ry * ry + rz * rz);
    const float e4[4] = {rx, ry, rz, dd};

    float e8[8];
#pragma unroll
    for (int o = 0; o < 8; ++o) {
      float s = 0.f;
#pragma unroll
      for (int i = 0; i < 4; ++i) s += e4[i] * sW[0 + o * 4 + i];
      e8[o] = fmaxf(s * (sW[32 + o] * BN_INV) + sW[40 + o], 0.f);
    }
    float e16[16];
#pragma unroll
    for (int o = 0; o < 16; ++o) {
      float s = 0.f;
#pragma unroll
      for (int i = 0; i < 8; ++i) s += e8[i] * sW[48 + o * 8 + i];
      e16[o] = fmaxf(s * (sW[176 + o] * BN_INV) + sW[192 + o], 0.f);
    }
    const float* nf = features + ((size_t)bb * NPTS + idx) * CIN;

    // A fragment, x = [nf(8) | e16(16) | pad(8)], K padded 24 -> 32.
    // lanes 0-15: K 0-7 & 16-23 of row r; lanes 16-31: K 8-15 & 24-31.
    v16h a1;
    if (hi == 0) {
#pragma unroll
      for (int i = 0; i < 8; ++i) a1[i] = (_Float16)nf[i];          // K 0..7
#pragma unroll
      for (int i = 0; i < 8; ++i) a1[8 + i] = (_Float16)e16[8 + i]; // K 16..23
    } else {
#pragma unroll
      for (int i = 0; i < 8; ++i) a1[i] = (_Float16)e16[i];         // K 8..15
#pragma unroll
      for (int i = 0; i < 8; ++i) a1[8 + i] = (_Float16)0.0f;       // pad
    }

    // m1: 24 -> 32, BN+ReLU, stage to LDS for layout transpose
#pragma unroll
    for (int t = 0; t < 2; ++t) {
      v8f h = wmma_f16(a1, Bm1[t], zero);
      const float g = sW[208 + t * 16 + col] * BN_INV;
      const float b = sW[240 + t * 16 + col];
#pragma unroll
      for (int i = 0; i < 8; ++i)
        sX[wave][m0 + i][t * 16 + col] = (_Float16)fmaxf(h[i] * g + b, 0.f);
    }
    __syncthreads();

    // A fragment for m2 (K = 32) from staged rows
    v16h a2;
#pragma unroll
    for (int i = 0; i < 8; ++i) {
      a2[i]     = sX[wave][r][hi * 8 + i];
      a2[8 + i] = sX[wave][r][16 + hi * 8 + i];
    }
    __syncthreads();

    // m2: 32 -> 64, BN+ReLU; keep f32 in regs (needed for pooling) + stage
    v8f y[4];
#pragma unroll
    for (int t = 0; t < 4; ++t) {
      v8f h = wmma_f16(a2, Bm2[t], zero);
      const float g = sW[272 + t * 16 + col] * BN_INV;
      const float b = sW[336 + t * 16 + col];
#pragma unroll
      for (int i = 0; i < 8; ++i) {
        const float v = fmaxf(h[i] * g + b, 0.f);
        y[t][i] = v;
        sX[wave][m0 + i][t * 16 + col] = (_Float16)v;
      }
    }
    __syncthreads();

    // A fragments for attention (K = 64, two 32-chunks)
    v16h aa0, aa1;
#pragma unroll
    for (int i = 0; i < 8; ++i) {
      aa0[i]     = sX[wave][r][hi * 8 + i];
      aa0[8 + i] = sX[wave][r][16 + hi * 8 + i];
      aa1[i]     = sX[wave][r][32 + hi * 8 + i];
      aa1[8 + i] = sX[wave][r][48 + hi * 8 + i];
    }

    // scores = x @ att_w^T; softmax over the 16 neighbor rows of each column
    // (8 in-lane rows + shfl_xor(16) to merge the two half-lanes), then
    // weighted pool into sPooled.
#pragma unroll
    for (int t = 0; t < 4; ++t) {
      v8f s = wmma_f16(aa0, Ba[0][t], zero);
      s     = wmma_f16(aa1, Ba[1][t], s);
      float mx = s[0];
#pragma unroll
      for (int i = 1; i < 8; ++i) mx = fmaxf(mx, s[i]);
      mx = fmaxf(mx, __shfl_xor(mx, 16, 32));
      float sum = 0.f, e[8];
#pragma unroll
      for (int i = 0; i < 8; ++i) { e[i] = __expf(s[i] - mx); sum += e[i]; }
      sum += __shfl_xor(sum, 16, 32);
      const float inv = 1.f / sum;
      float acc = 0.f;
#pragma unroll
      for (int i = 0; i < 8; ++i) acc += e[i] * y[t][i];
      acc *= inv;
      acc += __shfl_xor(acc, 16, 32);
      if (hi == 0) sPooled[wave * 4 + it][t * 16 + col] = acc;
    }
    __syncthreads();
  }

  // ---- phase 2: final 64->64 GEMM over the workgroup's 16 points --------
  if (wave == 0) {
    v16h pa0, pa1;
#pragma unroll
    for (int i = 0; i < 8; ++i) {
      pa0[i]     = (_Float16)sPooled[col][hi * 8 + i];
      pa0[8 + i] = (_Float16)sPooled[col][16 + hi * 8 + i];
      pa1[i]     = (_Float16)sPooled[col][32 + hi * 8 + i];
      pa1[8 + i] = (_Float16)sPooled[col][48 + hi * 8 + i];
    }
#pragma unroll
    for (int t = 0; t < 4; ++t) {
      const v16h b0 = load_btile(pool_w, 64, t * 16, 0,  lane);
      const v16h b1 = load_btile(pool_w, 64, t * 16, 32, lane);
      v8f o = wmma_f16(pa0, b0, zero);
      o     = wmma_f16(pa1, b1, o);
      const float g = sW[400 + t * 16 + col] * BN_INV;
      const float b = sW[464 + t * 16 + col];
#pragma unroll
      for (int i = 0; i < 8; ++i) {
        const float v  = fmaxf(o[i] * g + b, 0.f);
        const int   pt = blockIdx.x * 16 + m0 + i;
        out[(size_t)pt * COUT + t * 16 + col] = v;
      }
    }
  }
}

extern "C" void kernel_launch(void* const* d_in, const int* in_sizes, int n_in,
                              void* d_out, int out_size, void* d_ws, size_t ws_size,
                              hipStream_t stream) {
  (void)in_sizes; (void)n_in; (void)out_size; (void)d_ws; (void)ws_size;
  const float* coords   = (const float*)d_in[0];
  const float* features = (const float*)d_in[1];
  const int*   nbr      = (const int*)  d_in[2];
  const float* lse_w1   = (const float*)d_in[3];
  const float* lse_g1   = (const float*)d_in[4];
  const float* lse_b1   = (const float*)d_in[5];
  const float* lse_w2   = (const float*)d_in[6];
  const float* lse_g2   = (const float*)d_in[7];
  const float* lse_b2   = (const float*)d_in[8];
  const float* m1_w     = (const float*)d_in[9];
  const float* m1_g     = (const float*)d_in[10];
  const float* m1_b     = (const float*)d_in[11];
  const float* m2_w     = (const float*)d_in[12];
  const float* m2_g     = (const float*)d_in[13];
  const float* m2_b     = (const float*)d_in[14];
  const float* att_w    = (const float*)d_in[15];
  const float* pool_w   = (const float*)d_in[16];
  const float* pool_g   = (const float*)d_in[17];
  const float* pool_b   = (const float*)d_in[18];
  float* out = (float*)d_out;

  const int total_points = 2 * NPTS;        // B * N = 65536
  dim3 grid(total_points / 16);             // 16 points per workgroup
  dim3 block(128);                          // 4 waves * 32 lanes
  lfa_wmma_kernel<<<grid, block, 0, stream>>>(
      coords, features, nbr,
      lse_w1, lse_g1, lse_b1, lse_w2, lse_g2, lse_b2,
      m1_w, m1_g, m1_b, m2_w, m2_g, m2_b,
      att_w, pool_w, pool_g, pool_b, out);
}
